// SparseJL_8409545965787
// MI455X (gfx1250) — compile-verified, matrix operand order checked by reading, not run on hardware
//
#include <hip/hip_runtime.h>

// SparseJL forward: out = x @ Phi^T   (M=16384, N=1024, K=4096, fp32)
// bf16x3 emulated-fp32 GEMM on v_wmma_f32_16x16x32_bf16:
//   A = Ahi + Alo, B = Bhi + Blo (hi = top-16-bit truncation, exact; lo = bf16 residual)
//   D += Ahi*Bhi + Ahi*Blo + Alo*Bhi        (~3e-5 relative error)
// Round 3: modulo-3 fragment pipeline with three DISTINCT register sets and no
// rotation copies, so the allocator cannot coalesce them into one set (which in
// round 2 recreated the ds_load -> s_wait_dscnt 0 -> wmma serialization).

#define JL_IN_DIM 4096
#define JL_BM 128
#define JL_BN 128
#define JL_BK 32
#define JL_LDSS 40   // LDS row stride (ushort elems): 32 + 8 pad, keeps 16B align

typedef __attribute__((ext_vector_type(16))) __bf16          v16bf;
typedef __attribute__((ext_vector_type(8)))  float           v8f;
typedef __attribute__((ext_vector_type(8)))  unsigned short  v8u16;
typedef __attribute__((ext_vector_type(16))) unsigned short  v16u16;

__device__ __forceinline__ void jl_splitf(float f, unsigned short& hi, unsigned short& lo) {
  unsigned u = __float_as_uint(f);
  hi = (unsigned short)(u >> 16);                        // exact truncation to bf16
  float fh = __uint_as_float(u & 0xffff0000u);
  lo = (unsigned short)(__float_as_uint(f - fh) >> 16);  // residual -> bf16
}

__device__ __forceinline__ v16bf jl_frag(v8u16 a, v8u16 b) {
  return __builtin_bit_cast(
      v16bf, __builtin_shufflevector(a, b, 0, 1, 2, 3, 4, 5, 6, 7, 8, 9, 10, 11, 12, 13, 14, 15));
}

// 16 consecutive fp32
__device__ __forceinline__ void jl_ld16c(const float* __restrict__ p, float* v) {
  float4 a = ((const float4*)p)[0], b = ((const float4*)p)[1];
  float4 c = ((const float4*)p)[2], d = ((const float4*)p)[3];
  v[0] = a.x;  v[1] = a.y;  v[2] = a.z;  v[3] = a.w;
  v[4] = b.x;  v[5] = b.y;  v[6] = b.z;  v[7] = b.w;
  v[8] = c.x;  v[9] = c.y;  v[10] = c.z; v[11] = c.w;
  v[12] = d.x; v[13] = d.y; v[14] = d.z; v[15] = d.w;
}

// two 8-fp32 chunks at p and p+16 (WMMA 16-bit fragment K pattern)
__device__ __forceinline__ void jl_ld8x2(const float* __restrict__ p, float* v) {
  float4 a = ((const float4*)p)[0], b = ((const float4*)p)[1];
  float4 c = ((const float4*)(p + 16))[0], d = ((const float4*)(p + 16))[1];
  v[0] = a.x;  v[1] = a.y;  v[2] = a.z;  v[3] = a.w;
  v[4] = b.x;  v[5] = b.y;  v[6] = b.z;  v[7] = b.w;
  v[8] = c.x;  v[9] = c.y;  v[10] = c.z; v[11] = c.w;
  v[12] = d.x; v[13] = d.y; v[14] = d.z; v[15] = d.w;
}

__device__ __forceinline__ void jl_stage(const float* v, unsigned short* __restrict__ dh,
                                         unsigned short* __restrict__ dl) {
  v16u16 hv, lv;
#pragma unroll
  for (int j = 0; j < 16; ++j) {
    unsigned short th, tl;
    jl_splitf(v[j], th, tl);
    hv[j] = th; lv[j] = tl;
  }
  *(v8u16*)(dh)     = __builtin_shufflevector(hv, hv, 0, 1, 2, 3, 4, 5, 6, 7);
  *(v8u16*)(dh + 8) = __builtin_shufflevector(hv, hv, 8, 9, 10, 11, 12, 13, 14, 15);
  *(v8u16*)(dl)     = __builtin_shufflevector(lv, lv, 0, 1, 2, 3, 4, 5, 6, 7);
  *(v8u16*)(dl + 8) = __builtin_shufflevector(lv, lv, 8, 9, 10, 11, 12, 13, 14, 15);
}

__global__ __launch_bounds__(256) void sparsejl_wmma_kernel(
    const float* __restrict__ x, const float* __restrict__ phi,
    float* __restrict__ out, int out_dim) {
  __shared__ __align__(16) unsigned short Ahi[2][JL_BM * JL_LDSS];  // 2 x 10 KB
  __shared__ __align__(16) unsigned short Alo[2][JL_BM * JL_LDSS];  // 2 x 10 KB

  const int tid   = threadIdx.x;
  const int lane  = tid & 31;
  const int wave  = tid >> 5;      // 0..7 -> 16-wide N-subtile
  const int lrow  = lane & 15;
  const int lkgrp = lane >> 4;

  const size_t m0 = (size_t)blockIdx.y * JL_BM;
  const int    n0 = blockIdx.x * JL_BN + wave * 16;

  const float* bptr = phi + (size_t)(n0 + lrow) * JL_IN_DIM + (lkgrp << 3);

  // Cooperative A staging: 2 threads/row, 16 consecutive fp32 each
  const int arow  = tid >> 1;
  const int ahalf = (tid & 1) << 4;
  const float* aptr = x + (m0 + arow) * JL_IN_DIM + ahalf;
  const int stoff = arow * JL_LDSS + ahalf;

  v8f c[8];
#pragma unroll
  for (int i = 0; i < 8; ++i)
#pragma unroll
    for (int j = 0; j < 8; ++j) c[i][j] = 0.0f;

  // ---- prologue: stage k0=0 into buffer 0, fetch B(0) ----
  float av[16], bv[16];
  jl_ld16c(aptr, av);
  jl_ld8x2(bptr, bv);
  jl_stage(av, &Ahi[0][stoff], &Alo[0][stoff]);
  __syncthreads();

  const int abase0 = lrow * JL_LDSS + (lkgrp << 3);
  int cur = 0;

#pragma unroll 1
  for (int k0 = 0; k0 < JL_IN_DIM; k0 += JL_BK) {
    const bool has_next = (k0 + JL_BK) < JL_IN_DIM;

    // 1) issue next step's global loads first -> latency hides behind WMMAs
    float an[16], bn[16];
    if (has_next) {
      jl_ld16c(aptr + (k0 + JL_BK), an);
      jl_ld8x2(bptr + (k0 + JL_BK), bn);
    }

    // 2) build current B fragments
    v16u16 bhv, blv;
#pragma unroll
    for (int j = 0; j < 16; ++j) {
      unsigned short th, tl;
      jl_splitf(bv[j], th, tl);
      bhv[j] = th; blv[j] = tl;
    }
    v16bf Bhi = __builtin_bit_cast(v16bf, bhv);
    v16bf Blo = __builtin_bit_cast(v16bf, blv);

    // 3) compute: modulo-3 software pipeline, three distinct fragment sets,
    //    no inter-set copies -> loads for tile i+3 issue right after tile i's
    //    WMMAs, giving ~6 WMMAs of slack before each s_wait_dscnt.
    const unsigned short* __restrict__ AH = &Ahi[cur][0];
    const unsigned short* __restrict__ AL = &Alo[cur][0];

    v8u16 s0h0, s0h1, s0l0, s0l1;
    v8u16 s1h0, s1h1, s1l0, s1l1;
    v8u16 s2h0, s2h1, s2l0, s2l1;

#define JL_LDF(i, H0, H1, L0, L1)                  \
    {                                              \
      const int ab = abase0 + (i) * 16 * JL_LDSS;  \
      H0 = *(const v8u16*)&AH[ab];                 \
      H1 = *(const v8u16*)&AH[ab + 16];            \
      L0 = *(const v8u16*)&AL[ab];                 \
      L1 = *(const v8u16*)&AL[ab + 16];            \
    }

#define JL_TILE(i, H0, H1, L0, L1)                                                                         \
    {                                                                                                      \
      v16bf Af = jl_frag(H0, H1);                                                                          \
      v16bf Al = jl_frag(L0, L1);                                                                          \
      c[i] = __builtin_amdgcn_wmma_f32_16x16x32_bf16(false, Af, false, Bhi, (short)0, c[i], false, false); \
      c[i] = __builtin_amdgcn_wmma_f32_16x16x32_bf16(false, Af, false, Blo, (short)0, c[i], false, false); \
      c[i] = __builtin_amdgcn_wmma_f32_16x16x32_bf16(false, Al, false, Bhi, (short)0, c[i], false, false); \
    }

    JL_LDF(0, s0h0, s0h1, s0l0, s0l1);
    JL_LDF(1, s1h0, s1h1, s1l0, s1l1);
    JL_LDF(2, s2h0, s2h1, s2l0, s2l1);

    JL_TILE(0, s0h0, s0h1, s0l0, s0l1);
    JL_LDF(3, s0h0, s0h1, s0l0, s0l1);
    JL_TILE(1, s1h0, s1h1, s1l0, s1l1);
    JL_LDF(4, s1h0, s1h1, s1l0, s1l1);
    JL_TILE(2, s2h0, s2h1, s2l0, s2l1);
    JL_LDF(5, s2h0, s2h1, s2l0, s2l1);
    JL_TILE(3, s0h0, s0h1, s0l0, s0l1);
    JL_LDF(6, s0h0, s0h1, s0l0, s0l1);
    JL_TILE(4, s1h0, s1h1, s1l0, s1l1);
    JL_LDF(7, s1h0, s1h1, s1l0, s1l1);
    JL_TILE(5, s2h0, s2h1, s2l0, s2l1);
    JL_TILE(6, s0h0, s0h1, s0l0, s0l1);
    JL_TILE(7, s1h0, s1h1, s1l0, s1l1);

#undef JL_LDF
#undef JL_TILE

    // 4) stage next A tile into the alternate buffer, single barrier
    if (has_next) {
      const int nxt = cur ^ 1;
      jl_stage(an, &Ahi[nxt][stoff], &Alo[nxt][stoff]);
#pragma unroll
      for (int j = 0; j < 16; ++j) bv[j] = bn[j];
    }
    __syncthreads();
    cur ^= 1;
  }

  // ---- write back: VGPR j = rows j / j+8, lanes = N cols ----
  const int orow = lkgrp << 3;
  const int ncol = n0 + lrow;
#pragma unroll
  for (int i = 0; i < 8; ++i) {
    const size_t rbase = m0 + (size_t)i * 16 + orow;
#pragma unroll
    for (int j = 0; j < 8; ++j) {
      out[(rbase + j) * (size_t)out_dim + ncol] = c[i][j];
    }
  }
}

extern "C" void kernel_launch(void* const* d_in, const int* in_sizes, int n_in,
                              void* d_out, int out_size, void* d_ws, size_t ws_size,
                              hipStream_t stream) {
  const float* x   = (const float*)d_in[0];
  const float* phi = (const float*)d_in[1];
  float* out       = (float*)d_out;

  const int nrows   = in_sizes[0] / JL_IN_DIM;  // 16384
  const int out_dim = in_sizes[1] / JL_IN_DIM;  // 1024

  dim3 grid(out_dim / JL_BN, nrows / JL_BM);  // N-blocks fastest -> x rows L2-coadjacent
  dim3 block(256);
  sparsejl_wmma_kernel<<<grid, block, 0, stream>>>(x, phi, out, out_dim);
}
